// GATLayer_14774687498688
// MI455X (gfx1250) — compile-verified, matrix operand order checked by reading, not run on hardware
//
#include <hip/hip_runtime.h>
#include <stdint.h>

// Problem sizes (fixed by the reference)
static constexpr int NN   = 4096;
static constexpr int FINN = 256;
static constexpr int FOUT = 256;

// ---- trivial ext-vector types (avoid HIP struct-vectors inside unions) ----
typedef __bf16        v16bf __attribute__((ext_vector_type(16)));
typedef float         v8f   __attribute__((ext_vector_type(8)));
typedef float         v4f   __attribute__((ext_vector_type(4)));
typedef unsigned int  v4u   __attribute__((ext_vector_type(4)));
typedef int           v4i   __attribute__((ext_vector_type(4)));
typedef int           v8i   __attribute__((ext_vector_type(8)));

union FragU {                 // one WMMA 16-bit operand fragment (32 bytes / lane)
  v4u            q[2];
  unsigned short us[16];
  v16bf          v;
};

__device__ __forceinline__ unsigned short f2bf(float f) {  // RNE f32->bf16
  unsigned int u = __float_as_uint(f);
  u += 0x7FFFu + ((u >> 16) & 1u);
  return (unsigned short)(u >> 16);
}

// ---------------------------------------------------------------------------
// K0: h = x @ W^T + b   via v_wmma_f32_16x16x32_bf16 (one 16x16 tile / wave)
// ---------------------------------------------------------------------------
__global__ __launch_bounds__(256) void k_h_gemm(const float* __restrict__ x,
                                                const float* __restrict__ W,
                                                const float* __restrict__ b,
                                                float* __restrict__ h) {
  const int tid  = threadIdx.x;
  const int lane = tid & 31, wave = tid >> 5;
  const int half = lane >> 4, lr = lane & 15;
  const int m0 = blockIdx.x * 16;
  const int n0 = (blockIdx.y * 8 + wave) * 16;
  const int n  = n0 + lr;

  v8f acc;
  const float bn = b[n];
#pragma unroll
  for (int i = 0; i < 8; ++i) acc[i] = bn;

  const float* xrow = x + (size_t)(m0 + lr) * FINN;
  const float* wrow = W + (size_t)n * FINN;

#pragma unroll
  for (int kk = 0; kk < FINN; kk += 32) {
    FragU a, bb;
    v4f xa0 = *(const v4f*)(xrow + kk + 8 * half);
    v4f xa1 = *(const v4f*)(xrow + kk + 8 * half + 4);
    v4f xa2 = *(const v4f*)(xrow + kk + 16 + 8 * half);
    v4f xa3 = *(const v4f*)(xrow + kk + 16 + 8 * half + 4);
    v4f wb0 = *(const v4f*)(wrow + kk + 8 * half);
    v4f wb1 = *(const v4f*)(wrow + kk + 8 * half + 4);
    v4f wb2 = *(const v4f*)(wrow + kk + 16 + 8 * half);
    v4f wb3 = *(const v4f*)(wrow + kk + 16 + 8 * half + 4);
#pragma unroll
    for (int e = 0; e < 4; ++e) {
      a.us[e]      = f2bf(xa0[e]);  a.us[4 + e]  = f2bf(xa1[e]);
      a.us[8 + e]  = f2bf(xa2[e]);  a.us[12 + e] = f2bf(xa3[e]);
      bb.us[e]     = f2bf(wb0[e]);  bb.us[4 + e] = f2bf(wb1[e]);
      bb.us[8 + e] = f2bf(wb2[e]);  bb.us[12 + e]= f2bf(wb3[e]);
    }
    acc = __builtin_amdgcn_wmma_f32_16x16x32_bf16(false, a.v, false, bb.v,
                                                  (short)0, acc, false, false);
  }
#pragma unroll
  for (int v = 0; v < 8; ++v) {
    const int M = m0 + v + 8 * half;
    h[(size_t)M * FOUT + n] = acc[v];
  }
}

// ---------------------------------------------------------------------------
// K1: a1[i] = h[i,:].w1, a2[i] = h[i,:].w2   (one wave32 per row)
// ---------------------------------------------------------------------------
__global__ __launch_bounds__(256) void k_att_dots(const float* __restrict__ h,
                                                  const float* __restrict__ att_w,
                                                  float* __restrict__ a1,
                                                  float* __restrict__ a2) {
  const int tid = threadIdx.x;
  const int lane = tid & 31, wave = tid >> 5;
  const int row = blockIdx.x * 8 + wave;
  const float* hr = h + (size_t)row * FOUT + lane * 8;
  v4f h0 = *(const v4f*)hr,                 v4f_h1;
  v4f h1 = *(const v4f*)(hr + 4);
  v4f u0 = *(const v4f*)(att_w + lane * 8), u1 = *(const v4f*)(att_w + lane * 8 + 4);
  v4f w0 = *(const v4f*)(att_w + FOUT + lane * 8);
  v4f w1 = *(const v4f*)(att_w + FOUT + lane * 8 + 4);
  float d1 = 0.f, d2 = 0.f;
#pragma unroll
  for (int e = 0; e < 4; ++e) {
    d1 += h0[e] * u0[e] + h1[e] * u1[e];
    d2 += h0[e] * w0[e] + h1[e] * w1[e];
  }
#pragma unroll
  for (int off = 16; off > 0; off >>= 1) {
    d1 += __shfl_down(d1, off, 32);
    d2 += __shfl_down(d2, off, 32);
  }
  if (lane == 0) { a1[row] = d1; a2[row] = d2; }
}

// ---------------------------------------------------------------------------
// K2a: per-row online-softmax stats over edges
// ---------------------------------------------------------------------------
__global__ __launch_bounds__(256) void k_row_stats(const int* __restrict__ adj,
                                                   const float* __restrict__ a2,
                                                   float* __restrict__ rowmax,
                                                   float* __restrict__ rowsum,
                                                   int* __restrict__ cnt) {
  __shared__ float sm[256], ss[256];
  __shared__ int sc[256];
  const int i = blockIdx.x, t = threadIdx.x;
  const int* arow = adj + (size_t)i * NN;
  float m = -1e30f, s = 0.f;
  int c = 0;
  for (int j = t; j < NN; j += 256) {
    if (arow[j]) {
      const float v = a2[j];
      ++c;
      if (v > m) { s = s * __expf(m - v) + 1.0f; m = v; }
      else       { s += __expf(v - m); }
    }
  }
  sm[t] = m; ss[t] = s; sc[t] = c;
  __syncthreads();
  for (int off = 128; off > 0; off >>= 1) {
    if (t < off) {
      const float m2 = sm[t + off], s2 = ss[t + off];
      const float M = fmaxf(sm[t], m2);
      ss[t] = ss[t] * __expf(sm[t] - M) + s2 * __expf(m2 - M);
      sm[t] = M;
      sc[t] += sc[t + off];
    }
    __syncthreads();
  }
  if (t == 0) { rowmax[i] = sm[0]; rowsum[i] = ss[0]; cnt[i] = sc[0]; }
}

// ---------------------------------------------------------------------------
// K2b: gmax, denom and exclusive prefix of per-row edge counts
// ---------------------------------------------------------------------------
__global__ __launch_bounds__(256) void k_global_reduce(const float* __restrict__ a1,
                                                       const float* __restrict__ rowmax,
                                                       const float* __restrict__ rowsum,
                                                       const int* __restrict__ cnt,
                                                       float* __restrict__ scal,
                                                       int* __restrict__ rowstart) {
  __shared__ float red[256];
  const int t = threadIdx.x;
  float gm = -1e30f;
  for (int i = t; i < NN; i += 256) gm = fmaxf(gm, a1[i] + rowmax[i]);
  red[t] = gm; __syncthreads();
  for (int off = 128; off > 0; off >>= 1) {
    if (t < off) red[t] = fmaxf(red[t], red[t + off]);
    __syncthreads();
  }
  const float gmax = red[0];
  __syncthreads();
  float ds = 0.f;
  for (int i = t; i < NN; i += 256) ds += __expf(a1[i] + rowmax[i] - gmax) * rowsum[i];
  red[t] = ds; __syncthreads();
  for (int off = 128; off > 0; off >>= 1) {
    if (t < off) red[t] += red[t + off];
    __syncthreads();
  }
  if (t == 0) {
    scal[0] = gmax;
    scal[1] = red[0];
    int run = 0;
    for (int i = 0; i < NN; ++i) { rowstart[i] = run; run += cnt[i]; }
  }
}

// ---------------------------------------------------------------------------
// K2c: w_node[r] = softmax prob of r-th row-major edge, r < N
// ---------------------------------------------------------------------------
__global__ __launch_bounds__(256) void k_build_wnode(const int* __restrict__ adj,
                                                     const float* __restrict__ a1,
                                                     const float* __restrict__ a2,
                                                     const int* __restrict__ rowstart,
                                                     const float* __restrict__ scal,
                                                     float* __restrict__ w_node) {
  const int i = blockIdx.x, t = threadIdx.x;
  const int rs = rowstart[i];
  if (rs >= NN) return;
  __shared__ int sp[256];
  __shared__ int sbase;
  if (t == 0) sbase = 0;
  __syncthreads();
  const float gmax = scal[0], denom = scal[1];
  const float a1i = a1[i];
  const int* arow = adj + (size_t)i * NN;
  for (int cb = 0; cb < NN; cb += 256) {
    const int j = cb + t;
    const int pred = arow[j] ? 1 : 0;
    sp[t] = pred;
    __syncthreads();
    for (int off = 1; off < 256; off <<= 1) {      // Hillis-Steele inclusive scan
      const int v = (t >= off) ? sp[t - off] : 0;
      __syncthreads();
      sp[t] += v;
      __syncthreads();
    }
    const int incl = sp[t];
    const int total = sp[255];
    const int rank = rs + sbase + incl - pred;
    if (pred && rank < NN)
      w_node[rank] = __expf(a1i + a2[j] - gmax) / denom;
    __syncthreads();
    if (t == 0) sbase += total;
    __syncthreads();
    if (rs + sbase >= NN) break;
  }
}

// ---------------------------------------------------------------------------
// K2d: hwT[n][j] = bf16( w_node[j] * h[j][n] )  (64x64 LDS tile transpose)
// ---------------------------------------------------------------------------
__global__ __launch_bounds__(256) void k_build_hwT(const float* __restrict__ h,
                                                   const float* __restrict__ w_node,
                                                   unsigned short* __restrict__ hwT) {
  __shared__ unsigned short tile[64][65];
  const int t = threadIdx.x;
  const int j0 = blockIdx.x * 64, n0 = blockIdx.y * 64;
#pragma unroll
  for (int r = 0; r < 16; ++r) {
    const int jl = r * 4 + (t >> 6);
    const int nl = t & 63;
    const float v = w_node[j0 + jl] * h[(size_t)(j0 + jl) * FOUT + n0 + nl];
    tile[nl][jl] = f2bf(v);
  }
  __syncthreads();
#pragma unroll
  for (int r = 0; r < 16; ++r) {
    const int nl = r * 4 + (t >> 6);
    const int jl = t & 63;
    hwT[(size_t)(n0 + nl) * NN + j0 + jl] = tile[nl][jl];
  }
}

// ---------------------------------------------------------------------------
// K3: out = relu( adj_bf16 @ hwT^T )
// 512 threads = 16 waves; block owns 16 rows (M) x all 256 cols (N).
// Adjacency chunks (16 x 256 int32) are DMA'd into LDS by the Tensor Data
// Mover (TENSOR_LOAD_TO_LDS, tracked by TENSORcnt), double-buffered, with the
// D# pad feature (4 DWORDs per 256) giving a 260-int row pitch -> per-lane
// 16B fragment reads are bank-conflict-free. 0/1 -> bf16 conversion happens
// at fragment-assembly time (exact).
// ---------------------------------------------------------------------------
#define HAVE_TDM __has_builtin(__builtin_amdgcn_tensor_load_to_lds)

static constexpr int KCH    = 256;
static constexpr int APITCH = 260;   // ints per LDS row after TDM padding

__global__ __launch_bounds__(512) void k_aggregate(const int* __restrict__ adj,
                                                   const unsigned short* __restrict__ hwT,
                                                   float* __restrict__ out) {
  __shared__ __attribute__((aligned(16))) int ldsA[2][16 * APITCH + 8];
  const int tid  = threadIdx.x;
  const int lane = tid & 31, wave = tid >> 5;
  const int half = lane >> 4, lr = lane & 15;
  const int i0 = blockIdx.x * 16;
  const int n0 = wave * 16;

  v8f acc;
#pragma unroll
  for (int i = 0; i < 8; ++i) acc[i] = 0.f;

  const unsigned short* hrow = hwT + (size_t)(n0 + lr) * NN;

#if HAVE_TDM
  // --- Tensor DMA descriptor group1 (constant): 2D tile 256x16 of 4B data,
  //     tensor row stride 4096 elems, LDS pad 4 DW per 256 DW.
  v8i g1;
  g1[0] = (2 << 16)        // data_size = 4 bytes
        | (1 << 20)        // pad_enable
        | (7 << 22)        // pad_interval code 7 = 256 DWORDs
        | (3 << 25);       // pad_amount  code 3 = 4 DWORDs
  g1[1] = (int)((4096u & 0xFFFFu) << 16);          // tensor_dim0[15:0]
  g1[2] = (int)((4096u >> 16) | ((4096u & 0xFFFFu) << 16)); // td0 hi | td1 lo
  g1[3] = (int)((4096u >> 16) | (256u << 16));     // td1 hi | tile_dim0 = 256
  g1[4] = 16;                                      // tile_dim1 = 16, tile_dim2 = 0
  g1[5] = 4096;                                    // tensor_dim0_stride[31:0]
  g1[6] = 0;                                       // td0_stride hi | td1_stride lo
  g1[7] = 0;                                       // td1_stride hi
  const v4i gz4 = {0, 0, 0, 0};                    // groups 2/3 unused (2D)
  const v8i gz8 = {0, 0, 0, 0, 0, 0, 0, 0};        // extra 6-arg-form group

  const unsigned long long gbase =
      (unsigned long long)(uintptr_t)(adj + (size_t)i0 * NN);
  const unsigned ldsaddr[2] = {
      (unsigned)(uintptr_t)(&ldsA[0][0]),
      (unsigned)(uintptr_t)(&ldsA[1][0]) };

  auto tdm_issue = [&](int buf, int c) {
    const unsigned long long ga = gbase + (unsigned long long)c * (KCH * 4);
    v4u g0;
    g0[0] = 1u;                                         // count=1, user mode
    g0[1] = ldsaddr[buf];                               // lds_addr
    g0[2] = (unsigned)(ga & 0xFFFFFFFFull);             // global_addr lo
    g0[3] = (unsigned)((ga >> 32) & 0x1FFFFFFull) | (2u << 30);  // hi | type=2
    __builtin_amdgcn_tensor_load_to_lds(g0, g1, gz4, gz4, gz8, 0);
  };

  if (wave == 0) tdm_issue(0, 0);
#else
  const int lrow = tid >> 5;
  const int lk   = (tid & 31) * 8;
  const int* aptr = adj + (size_t)(i0 + lrow) * NN + lk;
  v4i r0 = *(const v4i*)(aptr);
  v4i r1 = *(const v4i*)(aptr + 4);
#endif

  for (int c = 0; c < NN / KCH; ++c) {
    const int buf = c & 1;
#if HAVE_TDM
    if (wave == 0) {
      if (c + 1 < NN / KCH) {
        tdm_issue(buf ^ 1, c + 1);                 // overlap next DMA
        __builtin_amdgcn_s_wait_tensorcnt(1);      // chunk c landed (in-order)
      } else {
        __builtin_amdgcn_s_wait_tensorcnt(0);
      }
    }
    __syncthreads();                               // publish LDS to all waves
#else
#pragma unroll
    for (int e = 0; e < 4; ++e) {
      ldsA[buf][lrow * APITCH + lk + e]     = r0[e];
      ldsA[buf][lrow * APITCH + lk + 4 + e] = r1[e];
    }
    __syncthreads();
    if (c + 1 < NN / KCH) {
      const int* np = aptr + (c + 1) * KCH;
      r0 = *(const v4i*)(np);
      r1 = *(const v4i*)(np + 4);
    }
#endif
    const int* lbase = &ldsA[buf][0];
#pragma unroll
    for (int s = 0; s < KCH / 32; ++s) {
      const int ks = s * 32;
      FragU a, bb;
      const int* ap = lbase + lr * APITCH + ks + 8 * half;
      v4i q0 = *(const v4i*)(ap);
      v4i q1 = *(const v4i*)(ap + 4);
      v4i q2 = *(const v4i*)(ap + 16);
      v4i q3 = *(const v4i*)(ap + 20);
#pragma unroll
      for (int e = 0; e < 4; ++e) {                // int 0/1 -> bf16 exactly
        a.us[e]      = q0[e] ? (unsigned short)0x3F80 : (unsigned short)0;
        a.us[4 + e]  = q1[e] ? (unsigned short)0x3F80 : (unsigned short)0;
        a.us[8 + e]  = q2[e] ? (unsigned short)0x3F80 : (unsigned short)0;
        a.us[12 + e] = q3[e] ? (unsigned short)0x3F80 : (unsigned short)0;
      }
      const unsigned short* hp = hrow + c * KCH + ks + 8 * half;
      bb.q[0] = *(const v4u*)(hp);
      bb.q[1] = *(const v4u*)(hp + 16);
      acc = __builtin_amdgcn_wmma_f32_16x16x32_bf16(false, a.v, false, bb.v,
                                                    (short)0, acc, false, false);
    }
    __syncthreads();                               // done reading buf before reuse
  }
#pragma unroll
  for (int v = 0; v < 8; ++v) {
    const int M = i0 + v + 8 * half;
    out[(size_t)M * FOUT + n0 + lr] = fmaxf(acc[v], 0.0f);
  }
}

// ---------------------------------------------------------------------------
extern "C" void kernel_launch(void* const* d_in, const int* in_sizes, int n_in,
                              void* d_out, int out_size, void* d_ws, size_t ws_size,
                              hipStream_t stream) {
  const float* x     = (const float*)d_in[0];
  const int*   adj   = (const int*)  d_in[1];
  const float* W     = (const float*)d_in[2];
  const float* b     = (const float*)d_in[3];
  const float* att_w = (const float*)d_in[4];
  // d_in[5] (att_b) cancels between exp(s-gmax) and denom -> unused.
  (void)in_sizes; (void)n_in; (void)out_size; (void)ws_size;

  char* ws = (char*)d_ws;
  float*          h        = (float*)(ws);                         // 4 MB
  unsigned short* hwT      = (unsigned short*)(ws + 4194304);      // 2 MB
  float*          a1       = (float*)(ws + 6291456);
  float*          a2       = (float*)(ws + 6307840);
  float*          rowmax   = (float*)(ws + 6324224);
  float*          rowsum   = (float*)(ws + 6340608);
  int*            cnt      = (int*)  (ws + 6356992);
  int*            rowstart = (int*)  (ws + 6373376);
  float*          w_node   = (float*)(ws + 6389760);
  float*          scal     = (float*)(ws + 6406144);
  float*          out      = (float*)d_out;

  k_h_gemm      <<<dim3(NN / 16, 2), 256, 0, stream>>>(x, W, b, h);
  k_att_dots    <<<NN / 8,           256, 0, stream>>>(h, att_w, a1, a2);
  k_row_stats   <<<NN,               256, 0, stream>>>(adj, a2, rowmax, rowsum, cnt);
  k_global_reduce<<<1,               256, 0, stream>>>(a1, rowmax, rowsum, cnt, scal, rowstart);
  k_build_wnode <<<NN,               256, 0, stream>>>(adj, a1, a2, rowstart, scal, w_node);
  k_build_hwT   <<<dim3(NN / 64, FOUT / 64), 256, 0, stream>>>(h, w_node, hwT);
  k_aggregate   <<<NN / 16,          512, 0, stream>>>(adj, hwT, out);
}
